// SparseMultiHeadAttention_67860483277139
// MI455X (gfx1250) — matrix operand order repaired and checked by reading, not compiled, not run on hardware
//
#include <hip/hip_runtime.h>
#include <hip/hip_bf16.h>
#include <stdint.h>

typedef __attribute__((ext_vector_type(16))) __bf16 v16bf;
typedef __attribute__((ext_vector_type(8)))  float  v8f;

namespace {
constexpr int Bsz = 2, Tt = 2048, Dd = 1024, Hh = 16, DHd = 64, TOPK = 32;
constexpr int Mrows = Bsz * Tt;   // 4096
}

__device__ __forceinline__ unsigned short f2bf(float f) {
  union { float f; unsigned u; } x; x.f = f;
  unsigned u = x.u;
  return (unsigned short)((u + 0x7FFFu + ((u >> 16) & 1u)) >> 16);  // RNE
}

// 32-bit LDS byte address of a generic pointer that points into shared memory
__device__ __forceinline__ unsigned lds_addr_of(const void* p) {
  return (unsigned)(unsigned long long)(__attribute__((address_space(3))) const char*)p;
}

// per-lane async copy of 16 bytes global -> LDS (ASYNCcnt-tracked)
__device__ __forceinline__ void async_g2l_b128(unsigned lds_byte_addr, const void* gsrc) {
  asm volatile("global_load_async_to_lds_b128 %0, %1, off"
               :: "v"(lds_byte_addr), "v"(gsrc) : "memory");
}

union FragAB { v16bf v; uint4 q[2]; };
union FragC  { v8f  v; float f[8]; };

// ---------------------------------------------------------------- fp32 -> bf16
__global__ __launch_bounds__(256) void smha_cvt_bf16_k(
    const float4* __restrict__ in, uint2* __restrict__ out, int n4) {
  int i = blockIdx.x * 256 + threadIdx.x;
  if (i < n4) {
    float4 f = in[i];
    uint2 o;
    o.x = (unsigned)f2bf(f.x) | ((unsigned)f2bf(f.y) << 16);
    o.y = (unsigned)f2bf(f.z) | ((unsigned)f2bf(f.w) << 16);
    out[i] = o;
  }
}

// ------------------------------------------------------------- bf16 WMMA GEMM
// C[m][n] = sum_k A[m][k] * W[n][k]   (A: 4096x1024, W: 1024x1024, both bf16 RM)
// mode 0: bf16 out, head-split [B,H,T,DH]            (Q, K)
// mode 1: bf16 out, head-split transposed [B,H,DH,T] (V^T for PV matmul)
// mode 2: fp32 out + bias, row-major [M, D]          (final projection)
__global__ __launch_bounds__(256) void smha_gemm_bf16_k(
    const unsigned short* __restrict__ A, const unsigned short* __restrict__ W,
    unsigned short* __restrict__ dst_bf, float* __restrict__ dst_f,
    const float* __restrict__ bias, int mode)
{
  constexpr int KD = Dd;
  const int gw   = blockIdx.x * 8 + (threadIdx.x >> 5);  // global wave id, 0..1023
  const int lane = threadIdx.x & 31;
  const int g = lane >> 4, r = lane & 15;
  const int mt = gw >> 4, nt = gw & 15;                  // 64 M-tiles x 16 N-tiles
  const int m0 = mt * 64, n0 = nt * 64;

  FragC c[4][4];
  #pragma unroll
  for (int i = 0; i < 4; ++i)
    #pragma unroll
    for (int j = 0; j < 4; ++j)
      #pragma unroll
      for (int e = 0; e < 8; ++e) c[i][j].f[e] = 0.f;

  for (int k0 = 0; k0 < KD; k0 += 32) {
    FragAB a[4];
    #pragma unroll
    for (int i = 0; i < 4; ++i) {
      const unsigned short* ar = A + (size_t)(m0 + i * 16 + r) * KD + k0;
      __builtin_prefetch(ar + 32, 0, 1);                 // global_prefetch_b8
      a[i].q[0] = *(const uint4*)(ar + g * 16);          // A: K = g*16..g*16+7
      a[i].q[1] = *(const uint4*)(ar + g * 16 + 8);      //    K = g*16+8..+15
    }
    #pragma unroll
    for (int j = 0; j < 4; ++j) {
      FragAB b;
      const unsigned short* wr = W + (size_t)(n0 + j * 16 + r) * KD + k0 + g * 16;
      b.q[0] = *(const uint4*)wr;                        // B: K = g*16..g*16+15
      b.q[1] = *(const uint4*)(wr + 8);
      #pragma unroll
      for (int i = 0; i < 4; ++i)
        c[i][j].v = __builtin_amdgcn_wmma_f32_16x16x32_bf16(
            false, a[i].v, false, b.v, (short)0, c[i][j].v, false, false);
    }
  }

  #pragma unroll
  for (int i = 0; i < 4; ++i) {
    #pragma unroll
    for (int j = 0; j < 4; ++j) {
      #pragma unroll
      for (int e = 0; e < 8; ++e) {
        const int m = m0 + i * 16 + e + 8 * g;           // C: M = e+8g, N = r
        const int n = n0 + j * 16 + r;
        const float v = c[i][j].f[e];
        if (mode == 2) {
          dst_f[(size_t)m * Dd + n] = v + bias[n];
        } else {
          const int bb = m >> 11, t = m & (Tt - 1);
          const int h = n >> 6, dh = n & (DHd - 1);
          const size_t idx = (mode == 0)
              ? (((size_t)(bb * Hh + h) * Tt + t) * DHd + dh)
              : (((size_t)(bb * Hh + h) * DHd + dh) * Tt + t);
          dst_bf[idx] = f2bf(v);
        }
      }
    }
  }
}

// --------------------------------------------- fused top-k attention per 16 rows
// block = (b*H + h) * (T/16) + qtile ; 256 threads (8 waves)
// LDS: S fp32[16][2048] (128K) | P bf16[16][2048] (64K) | Red fp32[4][256] (4K)
//      | Kbuf bf16 [8 waves][2 bufs][16x64] (32K, async-staged, per-wave)
__global__ __launch_bounds__(256) void smha_attn_topk_k(
    const unsigned short* __restrict__ Q, const unsigned short* __restrict__ K,
    const unsigned short* __restrict__ Vt, unsigned short* __restrict__ O)
{
  extern __shared__ char smem[];
  float*          S    = (float*)smem;
  unsigned short* P    = (unsigned short*)(smem + (size_t)16 * Tt * 4);
  float*          Red  = (float*)(smem + (size_t)16 * Tt * 4 + (size_t)16 * Tt * 2);
  unsigned short* Kbuf = (unsigned short*)(smem + (size_t)16 * Tt * 4 + (size_t)16 * Tt * 2 + 4096);

  const int qt = blockIdx.x & 127;
  const int bh = blockIdx.x >> 7;                 // b*H + h
  const int wave = threadIdx.x >> 5, lane = threadIdx.x & 31;
  const int g = lane >> 4, r = lane & 15;
  const size_t qkb = (size_t)bh * Tt * DHd;
  const size_t vb  = (size_t)bh * DHd * Tt;
  const int q0 = qt * 16;
  const float scale = 0.125f;                     // 1/sqrt(64)

  // Q A-fragments: loaded once, reused for all 128 key tiles
  FragAB a0, a1;
  {
    const unsigned short* qr = Q + qkb + (size_t)(q0 + r) * DHd;
    a0.q[0] = *(const uint4*)(qr + g * 16);      a0.q[1] = *(const uint4*)(qr + g * 16 + 8);
    a1.q[0] = *(const uint4*)(qr + 32 + g * 16); a1.q[1] = *(const uint4*)(qr + 40 + g * 16);
  }

  // ---- scores = Q K^T / sqrt(DH): K tiles async-staged into per-wave double buffer
  const unsigned short* gK = K + qkb;
  const unsigned kbase = lds_addr_of(Kbuf) + wave * 2 * 2048 + lane * 64; // bytes
  const int srow = lane >> 1, shalf = (lane & 1) * 32;  // lane stages 64B of the 2KB tile

  {           // prologue: stage tile for ntile = wave into buf 0
    const unsigned short* gsrc = gK + (size_t)(wave * 16 + srow) * DHd + shalf;
    async_g2l_b128(kbase +  0, gsrc);      async_g2l_b128(kbase + 16, gsrc +  8);
    async_g2l_b128(kbase + 32, gsrc + 16); async_g2l_b128(kbase + 48, gsrc + 24);
  }
  for (int i = 0; i < 16; ++i) {
    const int ntile = wave + 8 * i;
    const int cur = i & 1;
    if (i < 15) {    // stage next tile into the other buffer
      const unsigned short* gsrc = gK + (size_t)((ntile + 8) * 16 + srow) * DHd + shalf;
      const unsigned l = kbase + (cur ^ 1) * 2048;
      async_g2l_b128(l +  0, gsrc);      async_g2l_b128(l + 16, gsrc +  8);
      async_g2l_b128(l + 32, gsrc + 16); async_g2l_b128(l + 48, gsrc + 24);
      asm volatile("s_wait_asynccnt 0x4" ::: "memory");  // current tile done, next in flight
    } else {
      asm volatile("s_wait_asynccnt 0x0" ::: "memory");
    }
    const unsigned short* kr = Kbuf + (size_t)(wave * 2 + cur) * 1024 + r * 64;
    FragAB b0, b1;
    b0.q[0] = *(const uint4*)(kr + g * 16);       b0.q[1] = *(const uint4*)(kr + g * 16 + 8);
    b1.q[0] = *(const uint4*)(kr + 32 + g * 16);  b1.q[1] = *(const uint4*)(kr + 40 + g * 16);
    FragC c;
    #pragma unroll
    for (int e = 0; e < 8; ++e) c.f[e] = 0.f;
    c.v = __builtin_amdgcn_wmma_f32_16x16x32_bf16(false, a0.v, false, b0.v, (short)0, c.v, false, false);
    c.v = __builtin_amdgcn_wmma_f32_16x16x32_bf16(false, a1.v, false, b1.v, (short)0, c.v, false, false);
    #pragma unroll
    for (int e = 0; e < 8; ++e)
      S[(size_t)(e + 8 * g) * Tt + ntile * 16 + r] = c.f[e] * scale;
  }
  __syncthreads();

  // ---- per-row: 32nd-largest threshold (iterative max extraction) + softmax
  //      all passes use b64 (float2) LDS accesses
  for (int rr = 0; rr < 2; ++rr) {
    const int row = wave * 2 + rr;
    float* s = S + (size_t)row * Tt;
    float thr = 3.402823466e38f, rowmax = 0.f;
    for (int it = 0; it < TOPK; ++it) {
      float lm = -3.402823466e38f;
      for (int j = lane * 2; j < Tt; j += 64) {
        float2 v = *(const float2*)(s + j);
        if (v.x < thr && v.x > lm) lm = v.x;
        if (v.y < thr && v.y > lm) lm = v.y;
      }
      #pragma unroll
      for (int off = 16; off; off >>= 1) {
        float o = __shfl_xor(lm, off, 32);
        lm = (o > lm) ? o : lm;
      }
      if (it == 0) rowmax = lm;
      thr = lm;
    }
    float sum = 0.f;
    for (int j = lane * 2; j < Tt; j += 64) {
      float2 v = *(const float2*)(s + j);
      float px = (v.x >= thr) ? __expf(v.x - rowmax) : 0.f;
      float py = (v.y >= thr) ? __expf(v.y - rowmax) : 0.f;
      float2 o; o.x = px; o.y = py;
      *(float2*)(s + j) = o;
      sum += px + py;
    }
    #pragma unroll
    for (int off = 16; off; off >>= 1) sum += __shfl_xor(sum, off, 32);
    const float inv = 1.0f / sum;
    unsigned* prow = (unsigned*)(P + (size_t)row * Tt);
    for (int j = lane * 2; j < Tt; j += 64) {
      float2 v = *(const float2*)(s + j);
      prow[j >> 1] = (unsigned)f2bf(v.x * inv) | ((unsigned)f2bf(v.y * inv) << 16);
    }
  }
  __syncthreads();

  // ---- out = P @ V  (16 x 64); wave w: n-tile w&3, K-chunks split across pairs
  const int ntile = wave & 3;
  FragC acc;
  #pragma unroll
  for (int e = 0; e < 8; ++e) acc.f[e] = 0.f;
  for (int k0 = (wave >> 2) * 32; k0 < Tt; k0 += 64) {
    FragAB a, b;
    const unsigned short* pr = P + (size_t)r * Tt + k0;
    a.q[0] = *(const uint4*)(pr + g * 16);
    a.q[1] = *(const uint4*)(pr + g * 16 + 8);
    const unsigned short* vr = Vt + vb + (size_t)(ntile * 16 + r) * Tt + k0 + g * 16;
    b.q[0] = *(const uint4*)vr;
    b.q[1] = *(const uint4*)(vr + 8);
    acc.v = __builtin_amdgcn_wmma_f32_16x16x32_bf16(false, a.v, false, b.v, (short)0, acc.v, false, false);
  }
  if (wave >= 4) {
    #pragma unroll
    for (int e = 0; e < 8; ++e) Red[(wave - 4) * 256 + lane * 8 + e] = acc.f[e];
  }
  __syncthreads();
  if (wave < 4) {
    const int bb = bh / Hh, h = bh % Hh;
    #pragma unroll
    for (int e = 0; e < 8; ++e) {
      const float v = acc.f[e] + Red[wave * 256 + lane * 8 + e];
      const int t = q0 + e + 8 * g;
      const int dh = ntile * 16 + r;
      O[((size_t)bb * Tt + t) * Dd + h * DHd + dh] = f2bf(v);
    }
  }
}

// ------------------------------------------------------------------- launcher
extern "C" void kernel_launch(void* const* d_in, const int* in_sizes, int n_in,
                              void* d_out, int out_size, void* d_ws, size_t ws_size,
                              hipStream_t stream) {
  const float* x    = (const float*)d_in[0];
  const float* Wq   = (const float*)d_in[1];
  const float* Wk   = (const float*)d_in[2];
  const float* Wv   = (const float*)d_in[3];
  const float* Wo_w = (const float*)d_in[4];
  const float* Wo_b = (const float*)d_in[5];
  float* out = (float*)d_out;

  // workspace carve-up (~48 MB)
  char* p = (char*)d_ws;
  unsigned short* xb  = (unsigned short*)p; p += (size_t)Mrows * Dd * 2;  // 8 MB
  unsigned short* wqb = (unsigned short*)p; p += (size_t)Dd * Dd * 2;     // 2 MB
  unsigned short* wkb = (unsigned short*)p; p += (size_t)Dd * Dd * 2;
  unsigned short* wvb = (unsigned short*)p; p += (size_t)Dd * Dd * 2;
  unsigned short* wob = (unsigned short*)p; p += (size_t)Dd * Dd * 2;
  unsigned short* Qb  = (unsigned short*)p; p += (size_t)Mrows * Dd * 2;  // 8 MB
  unsigned short* Kb  = (unsigned short*)p; p += (size_t)Mrows * Dd * 2;
  unsigned short* Vtb = (unsigned short*)p; p += (size_t)Mrows * Dd * 2;
  unsigned short* AO  = (unsigned short*)p; p += (size_t)Mrows * Dd * 2;

  // 1) fp32 -> bf16 conversions
  const int nx4 = Mrows * Dd / 4, nw4 = Dd * Dd / 4;
  smha_cvt_bf16_k<<<(nx4 + 255) / 256, 256, 0, stream>>>((const float4*)x,    (uint2*)xb,  nx4);
  smha_cvt_bf16_k<<<(nw4 + 255) / 256, 256, 0, stream>>>((const float4*)Wq,   (uint2*)wqb, nw4);
  smha_cvt_bf16_k<<<(nw4 + 255) / 256, 256, 0, stream>>>((const float4*)Wk,   (uint2*)wkb, nw4);
  smha_cvt_bf16_k<<<(nw4 + 255) / 256, 256, 0, stream>>>((const float4*)Wv,   (uint2*)wvb, nw4);
  smha_cvt_bf16_k<<<(nw4 + 255) / 256, 256, 0, stream>>>((const float4*)Wo_w, (uint2*)wob, nw4);

  // 2) QKV projections (V written transposed per head)
  const int gblocks = (64 * 16) / 8;  // 1024 wave-tiles / 8 waves per block
  smha_gemm_bf16_k<<<gblocks, 256, 0, stream>>>(xb, wqb, Qb,  nullptr, nullptr, 0);
  smha_gemm_bf16_k<<<gblocks, 256, 0, stream>>>(xb, wkb, Kb,  nullptr, nullptr, 0);
  smha_gemm_bf16_k<<<gblocks, 256, 0, stream>>>(xb, wvb, Vtb, nullptr, nullptr, 1);

  // 3) fused top-k attention; scores live entirely in LDS (228 KB / WGP)
  const size_t smem = (size_t)16 * Tt * 4 + (size_t)16 * Tt * 2 + 4096 + 32768;
  hipFuncSetAttribute(reinterpret_cast<const void*>(smha_attn_topk_k),
                      hipFuncAttributeMaxDynamicSharedMemorySize, (int)smem);
  smha_attn_topk_k<<<Bsz * Hh * (Tt / 16), 256, smem, stream>>>(Qb, Kb, Vtb, AO);

  // 4) output projection with bias (fp32 out)
  smha_gemm_bf16_k<<<gblocks, 256, 0, stream>>>(AO, wob, nullptr, out, Wo_b, 2);
}
